// RobustSparseHalfspaceLearner_58162447122471
// MI455X (gfx1250) — compile-verified, matrix operand order checked by reading, not run on hardware
//
#include <hip/hip_runtime.h>
#include <cstdint>

#define N_SAMP   200
#define N_FEAT   16
#define S_PAIRS  19900                 // C(200,2)
#define F_PAIRS  120                   // C(16,2)
#define K_TOTAL  (F_PAIRS * S_PAIRS)   // 2,388,000
#define ROWS_PAD 208                   // 13 tiles of 16
#define MARGIN   0.1f

typedef __attribute__((ext_vector_type(2)))  float    v2f;
typedef __attribute__((ext_vector_type(8)))  float    v8f;
typedef __attribute__((ext_vector_type(16))) _Float16 v16h;

// ---------------------------------------------------------------------------
// Kernel 1: lf = diag(y) @ features, one 16-row tile per 32-thread wave.
// Full-precision f32 WMMA: 16x16 result accumulated from four 16x16x4 steps.
// A_c (16x4) is the K-slice [4c..4c+3] of diag(y); B_c (4x16) the matching
// feature rows. EXEC is all-ones (single full wave, branch-free selects).
// ---------------------------------------------------------------------------
__global__ __launch_bounds__(32)
void lf_wmma_kernel(const int* __restrict__ labels,
                    const float* __restrict__ feats,
                    float* __restrict__ lf_ws) {
  const int tile = blockIdx.x;     // 0..12
  const int lane = threadIdx.x;    // 0..31
  const int half = lane >> 4;      // 0 | 1
  const int m    = lane & 15;

  const int gm = tile * 16 + m;
  const float ym = (gm < N_SAMP) ? (2.0f * (float)labels[gm] - 1.0f) : 0.0f;

  v8f acc = {};
#if __has_builtin(__builtin_amdgcn_wmma_f32_16x16x4_f32)
  #pragma unroll
  for (int c = 0; c < 4; ++c) {
    // A layout (32-bit 16x4): lanes 0-15 hold K=0,1; lanes 16-31 hold K=2,3.
    const int k0 = 4 * c + 2 * half;           // global K of this lane's A.x
    v2f a;
    a.x = (m == k0    ) ? ym : 0.0f;
    a.y = (m == k0 + 1) ? ym : 0.0f;
    // B (4x16): VGPR0 -> K = half, VGPR1 -> K = 2+half; N = m.
    const int r0 = tile * 16 + 4 * c + half;
    const int r1 = tile * 16 + 4 * c + 2 + half;
    v2f b;
    b.x = (r0 < N_SAMP) ? feats[r0 * N_FEAT + m] : 0.0f;
    b.y = (r1 < N_SAMP) ? feats[r1 * N_FEAT + m] : 0.0f;
    acc = __builtin_amdgcn_wmma_f32_16x16x4_f32(
        /*neg_a=*/false, a, /*neg_b=*/false, b,
        /*c_mod=*/(short)0, acc, /*reuse_a=*/false, /*reuse_b=*/false);
  }
#else
  // Fallback (codegen-confirmed builtin): diag(y) is exact in f16; use only
  // K=0..15 of the 16x16x32 shape, zero-padding the rest.
  {
    v16h A, B;
    const _Float16 yh = (_Float16)ym;
    #pragma unroll
    for (int e = 0; e < 16; ++e) {
      const int K = 8 * half + (e & 7) + ((e >= 8) ? 16 : 0);
      A[e] = (K == m) ? yh : (_Float16)0;
      const int rr = tile * 16 + K;
      B[e] = (K < 16 && rr < N_SAMP) ? (_Float16)feats[rr * N_FEAT + m]
                                     : (_Float16)0;
    }
    acc = __builtin_amdgcn_wmma_f32_16x16x32_f16(
        false, A, false, B, (short)0, acc, false, false);
  }
#endif

  // C/D layout: VGPR r -> row (r + 8*half), col m. ws is padded to 208 rows.
  #pragma unroll
  for (int r = 0; r < 8; ++r)
    lf_ws[(tile * 16 + r + 8 * half) * N_FEAT + m] = acc[r];
}

// ---------------------------------------------------------------------------
// Kernel 2: one thread per (feature-pair, sample-pair). Store-BW bound:
// lf + y staged in LDS (14.1 KB), 8-byte vector stores for w/rows/cols.
// ---------------------------------------------------------------------------
__global__ __launch_bounds__(256)
void solve_kernel(const int* __restrict__ labels,
                  const float* __restrict__ lf_ws,
                  float* __restrict__ w_out,
                  int* __restrict__ rows_out,
                  int* __restrict__ cols_out) {
  __shared__ float s_lf[ROWS_PAD * N_FEAT];
  __shared__ float s_y[ROWS_PAD];
  const int tid = threadIdx.x;

  for (int idx = tid; idx < ROWS_PAD * N_FEAT; idx += 256) s_lf[idx] = lf_ws[idx];
  for (int idx = tid; idx < ROWS_PAD; idx += 256)
    s_y[idx] = (idx < N_SAMP) ? (2.0f * (float)labels[idx] - 1.0f) : 0.0f;
  __syncthreads();

  const int k = blockIdx.x * 256 + tid;
  if (k >= K_TOTAL) return;

  const int f = k / S_PAIRS;
  const int p = k - f * S_PAIRS;

  // p -> (i,j) over N=200: pairs before row i = i*(399-i)/2
  int i = (int)(0.5f * (399.0f - sqrtf(159201.0f - 8.0f * (float)p)));
  i = i < 0 ? 0 : (i > 198 ? 198 : i);
  while ((i + 1) * (399 - (i + 1)) / 2 <= p) ++i;
  while (i * (399 - i) / 2 > p) --i;
  const int j = p - i * (399 - i) / 2 + i + 1;

  // f -> (a,b) over D=16: pairs before row a = a*(31-a)/2
  int a = (int)(0.5f * (31.0f - sqrtf(961.0f - 8.0f * (float)f)));
  a = a < 0 ? 0 : (a > 14 ? 14 : a);
  while ((a + 1) * (31 - (a + 1)) / 2 <= f) ++a;
  while (a * (31 - a) / 2 > f) --a;
  const int b = f - a * (31 - a) / 2 + a + 1;

  const float M00 = s_lf[i * N_FEAT + a], M01 = s_lf[i * N_FEAT + b];
  const float M10 = s_lf[j * N_FEAT + a], M11 = s_lf[j * N_FEAT + b];
  const float r0 = s_y[i] - MARGIN, r1 = s_y[j] - MARGIN;

  // 2x2 pseudoinverse solve.
  const float det  = M00 * M11 - M01 * M10;
  const float frob = M00 * M00 + M01 * M01 + M10 * M10 + M11 * M11;
  float w0, w1;
  if (det * det > 1e-12f * frob * frob) {          // full rank: adj/det
    const float inv = 1.0f / det;
    w0 = ( M11 * r0 - M01 * r1) * inv;
    w1 = (-M10 * r0 + M00 * r1) * inv;
  } else if (frob > 0.0f) {                        // rank-1: M^T / ||M||_F^2
    const float inv = 1.0f / frob;
    w0 = (M00 * r0 + M10 * r1) * inv;
    w1 = (M01 * r0 + M11 * r1) * inv;
  } else {                                         // zero matrix
    w0 = 0.0f; w1 = 0.0f;
  }

  ((float2*)w_out)[k]  = make_float2(w0, w1);
  ((int2*)rows_out)[k] = make_int2(k, k);
  ((int2*)cols_out)[k] = make_int2(a, b);
}

// ---------------------------------------------------------------------------
extern "C" void kernel_launch(void* const* d_in, const int* in_sizes, int n_in,
                              void* d_out, int out_size, void* d_ws, size_t ws_size,
                              hipStream_t stream) {
  const int*   labels = (const int*)d_in[0];
  const float* feats  = (const float*)d_in[1];
  // d_in[2], d_in[3]: empty prev_* index arrays (unused)

  float* lf_ws = (float*)d_ws;                       // 208*16 f32 = 13,312 B

  float* w_out    = (float*)d_out;                   // 2*K f32
  int*   rows_out = (int*)d_out + 2 * (size_t)K_TOTAL;
  int*   cols_out = rows_out + 2 * (size_t)K_TOTAL;

  lf_wmma_kernel<<<13, 32, 0, stream>>>(labels, feats, lf_ws);

  const int nblk = (K_TOTAL + 255) / 256;
  solve_kernel<<<nblk, 256, 0, stream>>>(labels, lf_ws, w_out, rows_out, cols_out);
}
